// MambaLayer_6313601925269
// MI455X (gfx1250) — matrix operand order repaired and checked by reading, not compiled, not run on hardware
//
#include <hip/hip_runtime.h>
#include <hip/hip_bf16.h>

typedef __attribute__((ext_vector_type(16))) _Float16 v16h;
typedef __attribute__((ext_vector_type(8)))  _Float16 v8h;
typedef __attribute__((ext_vector_type(8)))  float    v8f;

#define LSEQ 4096
#define DMODEL 256
#define DIN 512
#define NSTATE 16
#define NCHUNK 64
#define CHLEN 64

// ---------------- workspace layout (bytes) ----------------
enum : size_t {
  oXN16  = 0,                                   // L*256 f16
  oWIN16 = oXN16  + (size_t)LSEQ*DMODEL*2,      // 1024*256 f16
  oWX16  = oWIN16 + (size_t)1024*256*2,         // 48*512 f16
  oWOUT16= oWX16  + (size_t)48*512*2,           // 256*512 f16
  oWO16  = oWOUT16+ (size_t)256*512*2,          // 256*512 f16
  oYCAT  = oWO16  + (size_t)256*512*2,          // L*512 f16
  oXZ    = oYCAT  + (size_t)LSEQ*512*2,         // L*1024 f32
  oXIS   = oXZ    + (size_t)LSEQ*1024*4,        // L*512 f32
  oXI16  = oXIS   + (size_t)LSEQ*512*4,         // L*512 f16
  oXDBL  = oXI16  + (size_t)LSEQ*512*2,         // L*48 f32
  oDELTA = oXDBL  + (size_t)LSEQ*48*4,          // L*512 f32
  oU     = oDELTA + (size_t)LSEQ*512*4,         // L*512 f32
  oACH   = oU     + (size_t)LSEQ*512*4,         // 8192*64 f32
  oBCH   = oACH   + (size_t)8192*64*4,
  oHIN   = oBCH   + (size_t)8192*64*4,
  oYCORE = oHIN   + (size_t)8192*64*4,          // L*512 f32
  oY16   = oYCORE + (size_t)LSEQ*512*4,         // L*512 f16
  oEND   = oY16   + (size_t)LSEQ*512*2
};

// ---------------- single 16x16 WMMA tile (used for the small N=48 GEMM) ----
__device__ __forceinline__ v8f wmma_tile(const _Float16* __restrict__ A,
                                         const _Float16* __restrict__ B,
                                         int K, int rowA, int colB) {
  const int lane = threadIdx.x & 31;
  const int hi   = lane >> 4;
  const _Float16* ap = A + (size_t)rowA * K + hi * 16;
  const _Float16* bp = B + (size_t)colB * K + hi * 16;
  v8f acc = {};
  for (int k = 0; k < K; k += 32) {
    v8h a0 = *(const v8h*)(ap + k);
    v8h a1 = *(const v8h*)(ap + k + 8);
    v16h a = __builtin_shufflevector(a0, a1, 0,1,2,3,4,5,6,7,8,9,10,11,12,13,14,15);
    v16h b = *(const v16h*)(bp + k);
    acc = __builtin_amdgcn_wmma_f32_16x16x32_f16(false, a, false, b,
                                                 (short)0, acc, false, false);
  }
  return acc;
}

// ---------------- register-blocked 32x64 tile: 2 M-tiles x 4 N-tiles --------
// A fragments reused across 4 WMMAs, B fragments across 2 -> 1.5 b128/WMMA
__device__ __forceinline__ void wmma_block_2x4(const _Float16* __restrict__ A,
                                               const _Float16* __restrict__ B,
                                               int K, int arow0, int arow1,
                                               int ncol0, v8f (&acc)[2][4]) {
  const int lane = threadIdx.x & 31;
  const int i15  = lane & 15;
  const int hi   = lane >> 4;
  const _Float16* ap0 = A + (size_t)arow0 * K + hi * 16;
  const _Float16* ap1 = A + (size_t)arow1 * K + hi * 16;
  const _Float16* bp[4];
  #pragma unroll
  for (int j = 0; j < 4; ++j)
    bp[j] = B + (size_t)(ncol0 + j * 16 + i15) * K + hi * 16;
  for (int k = 0; k < K; k += 32) {
    v16h a[2], b[4];
    {
      v8h x0 = *(const v8h*)(ap0 + k);
      v8h x1 = *(const v8h*)(ap0 + k + 8);
      a[0] = __builtin_shufflevector(x0, x1, 0,1,2,3,4,5,6,7,8,9,10,11,12,13,14,15);
      v8h y0 = *(const v8h*)(ap1 + k);
      v8h y1 = *(const v8h*)(ap1 + k + 8);
      a[1] = __builtin_shufflevector(y0, y1, 0,1,2,3,4,5,6,7,8,9,10,11,12,13,14,15);
    }
    #pragma unroll
    for (int j = 0; j < 4; ++j) b[j] = *(const v16h*)(bp[j] + k);
    #pragma unroll
    for (int mi = 0; mi < 2; ++mi)
      #pragma unroll
      for (int j = 0; j < 4; ++j)
        acc[mi][j] = __builtin_amdgcn_wmma_f32_16x16x32_f16(
            false, a[mi], false, b[j], (short)0, acc[mi][j], false, false);
  }
}

// ---------------- kernels ----------------
__global__ __launch_bounds__(256) void k_ln(const float* __restrict__ x,
                                            const float* __restrict__ gamma,
                                            const float* __restrict__ beta,
                                            _Float16* __restrict__ xn) {
  int l = blockIdx.x * blockDim.x + threadIdx.x;
  float s = 0.f, s2 = 0.f;
  for (int c = 0; c < DMODEL; ++c) {
    float v = x[(size_t)c * LSEQ + l];
    s += v; s2 += v * v;
  }
  float mu = s * (1.f / DMODEL);
  float var = s2 * (1.f / DMODEL) - mu * mu;
  float rs = rsqrtf(var + 1e-5f);
  for (int c = 0; c < DMODEL; ++c) {
    float v = (x[(size_t)c * LSEQ + l] - mu) * rs * gamma[c] + beta[c];
    xn[(size_t)l * DMODEL + c] = (_Float16)v;
  }
}

__global__ __launch_bounds__(256) void k_cvt(const float* __restrict__ s,
                                             _Float16* __restrict__ d, int n) {
  int i = blockIdx.x * blockDim.x + threadIdx.x;
  if (i < n) d[i] = (_Float16)s[i];
}

// GEMM1: xz = xn(rev?) * Win^T  (M=4096, N=1024, K=256), blocked 32x64
__global__ __launch_bounds__(256) void k_gemm1(const _Float16* __restrict__ A,
                                               const _Float16* __restrict__ B,
                                               float* __restrict__ D, int revA) {
  const int NB = 1024 / 64;                       // 16 N-blocks
  int wave = (blockIdx.x * blockDim.x + threadIdx.x) >> 5;
  if (wave >= (LSEQ / 32) * NB) return;           // wave-uniform
  int bm = wave / NB, bn = wave - bm * NB;
  int lane = threadIdx.x & 31, i15 = lane & 15, hi = lane >> 4;
  int r0 = bm * 32 + i15, r1 = r0 + 16;
  int a0 = revA ? (LSEQ - 1 - r0) : r0;
  int a1 = revA ? (LSEQ - 1 - r1) : r1;
  v8f acc[2][4] = {};
  wmma_block_2x4(A, B, 256, a0, a1, bn * 64, acc);
  #pragma unroll
  for (int mi = 0; mi < 2; ++mi)
    #pragma unroll
    for (int j = 0; j < 4; ++j)
      #pragma unroll
      for (int r = 0; r < 8; ++r)
        D[(size_t)(bm * 32 + mi * 16 + r + 8 * hi) * 1024 + bn * 64 + j * 16 + i15]
            = acc[mi][j][r];
}

// small GEMM: x_dbl = xi * Wx^T (M=4096, N=48, K=512), 1 tile per wave
__global__ __launch_bounds__(256) void k_gemm_xdbl(const _Float16* __restrict__ A,
                                                   const _Float16* __restrict__ B,
                                                   float* __restrict__ D) {
  int wave = (blockIdx.x * blockDim.x + threadIdx.x) >> 5;
  if (wave >= (LSEQ >> 4) * 3) return;
  int tm = wave / 3, tn = wave - tm * 3;
  int lane = threadIdx.x & 31, i15 = lane & 15, hi = lane >> 4;
  v8f acc = wmma_tile(A, B, 512, tm * 16 + i15, tn * 16 + i15);
  #pragma unroll
  for (int r = 0; r < 8; ++r)
    D[(size_t)(tm * 16 + r + 8 * hi) * 48 + tn * 16 + i15] = acc[r];
}

// branch output -> concat buffer (f16), output-row reversal for bwd branch
__global__ __launch_bounds__(256) void k_gemm_ycat(const _Float16* __restrict__ A,
                                                   const _Float16* __restrict__ B,
                                                   _Float16* __restrict__ Yc,
                                                   int rev, int colOff) {
  const int NB = 256 / 64;                        // 4 N-blocks
  int wave = (blockIdx.x * blockDim.x + threadIdx.x) >> 5;
  if (wave >= (LSEQ / 32) * NB) return;
  int bm = wave / NB, bn = wave - bm * NB;
  int lane = threadIdx.x & 31, i15 = lane & 15, hi = lane >> 4;
  v8f acc[2][4] = {};
  wmma_block_2x4(A, B, DIN, bm * 32 + i15, bm * 32 + 16 + i15, bn * 64, acc);
  #pragma unroll
  for (int mi = 0; mi < 2; ++mi)
    #pragma unroll
    for (int j = 0; j < 4; ++j)
      #pragma unroll
      for (int r = 0; r < 8; ++r) {
        int m = bm * 32 + mi * 16 + r + 8 * hi;
        int orow = rev ? (LSEQ - 1 - m) : m;
        Yc[(size_t)orow * 512 + colOff + bn * 64 + j * 16 + i15] = (_Float16)acc[mi][j][r];
      }
}

// final GEMM: out = ycat * Wo^T + bo, stored transposed (c, l)
__global__ __launch_bounds__(256) void k_gemm_final(const _Float16* __restrict__ A,
                                                    const _Float16* __restrict__ B,
                                                    const float* __restrict__ bias,
                                                    float* __restrict__ Out) {
  const int NB = 256 / 64;
  int wave = (blockIdx.x * blockDim.x + threadIdx.x) >> 5;
  if (wave >= (LSEQ / 32) * NB) return;
  int bm = wave / NB, bn = wave - bm * NB;
  int lane = threadIdx.x & 31, i15 = lane & 15, hi = lane >> 4;
  v8f acc[2][4] = {};
  wmma_block_2x4(A, B, 512, bm * 32 + i15, bm * 32 + 16 + i15, bn * 64, acc);
  #pragma unroll
  for (int mi = 0; mi < 2; ++mi)
    #pragma unroll
    for (int j = 0; j < 4; ++j) {
      int col = bn * 64 + j * 16 + i15;
      float bb = bias[col];
      #pragma unroll
      for (int r = 0; r < 8; ++r)
        Out[(size_t)col * LSEQ + bm * 32 + mi * 16 + r + 8 * hi] = acc[mi][j][r] + bb;
    }
}

// causal depthwise conv (k=4) + SiLU on xi half of xz
__global__ __launch_bounds__(256) void k_conv(const float* __restrict__ xz,
                                              const float* __restrict__ convw,
                                              const float* __restrict__ convb,
                                              float* __restrict__ xis,
                                              _Float16* __restrict__ xi16) {
  int g = blockIdx.x * blockDim.x + threadIdx.x;
  int d = g & (DIN - 1);
  int l = g >> 9;
  const float* w = convw + d * 4;
  float acc = convb[d];
  for (int j = 0; j < 4; ++j) {
    int ls = l - 3 + j;
    if (ls >= 0) acc = fmaf(w[j], xz[(size_t)ls * 1024 + d], acc);
  }
  float sl = acc / (1.f + __expf(-acc));
  xis[g] = sl;
  xi16[g] = (_Float16)sl;
}

// delta = softplus(dt * Wdt^T + bdt), u = delta * xi
__global__ __launch_bounds__(256) void k_delta_u(const float* __restrict__ xdbl,
                                                 const float* __restrict__ Wdt,
                                                 const float* __restrict__ bdt,
                                                 const float* __restrict__ xis,
                                                 float* __restrict__ delta,
                                                 float* __restrict__ u) {
  int g = blockIdx.x * blockDim.x + threadIdx.x;
  int d = g & (DIN - 1);
  int l = g >> 9;
  const float* xr = xdbl + (size_t)l * 48;
  const float* wr = Wdt + d * 16;
  float s = bdt[d];
  for (int r = 0; r < 16; ++r) s = fmaf(xr[r], wr[r], s);
  float dl = (s > 20.f) ? s : log1pf(__expf(s));
  delta[g] = dl;
  u[g] = dl * xis[g];
}

// pass 1: per-(chain, chunk) local composition (prodA, sumB)
__global__ __launch_bounds__(256) void k_scan1(const float* __restrict__ delta,
                                               const float* __restrict__ u,
                                               const float* __restrict__ xdbl,
                                               const float* __restrict__ Alog,
                                               float* __restrict__ ach,
                                               float* __restrict__ bch) {
  int g = blockIdx.x * blockDim.x + threadIdx.x;
  int s = g & 15;
  int d = (g >> 4) & (DIN - 1);
  int c = g >> 13;
  int chain = g & 8191;
  float Av = -__expf(Alog[d * 16 + s]);
  float ap = 1.f, bs = 0.f;
  int l0 = c * CHLEN;
  for (int i = 0; i < CHLEN; ++i) {
    int l = l0 + i;
    if (i + 16 < CHLEN) {                                 // global_prefetch_b8
      __builtin_prefetch(&delta[(size_t)(l + 16) * 512 + d], 0, 0);
      __builtin_prefetch(&u[(size_t)(l + 16) * 512 + d], 0, 0);
    }
    float dl = delta[(size_t)l * 512 + d];
    float dA = __expf(dl * Av);
    float dBu = u[(size_t)l * 512 + d] * xdbl[(size_t)l * 48 + 16 + s];
    bs = fmaf(dA, bs, dBu);
    ap *= dA;
  }
  ach[(size_t)c * 8192 + chain] = ap;
  bch[(size_t)c * 8192 + chain] = bs;
}

// pass 2: carry propagation across 64 chunks per chain
__global__ __launch_bounds__(256) void k_scan2(const float* __restrict__ ach,
                                               const float* __restrict__ bch,
                                               float* __restrict__ hin) {
  int chain = blockIdx.x * blockDim.x + threadIdx.x;
  float h = 0.f;
  for (int c = 0; c < NCHUNK; ++c) {
    hin[(size_t)c * 8192 + chain] = h;
    h = fmaf(ach[(size_t)c * 8192 + chain], h, bch[(size_t)c * 8192 + chain]);
  }
}

// pass 3: re-run chunks with carry, reduce over 16 states via wave32 shfl_xor
__global__ __launch_bounds__(256) void k_scan3(const float* __restrict__ delta,
                                               const float* __restrict__ u,
                                               const float* __restrict__ xdbl,
                                               const float* __restrict__ Alog,
                                               const float* __restrict__ hin,
                                               float* __restrict__ ycore) {
  int g = blockIdx.x * blockDim.x + threadIdx.x;
  int s = g & 15;
  int d = (g >> 4) & (DIN - 1);
  int c = g >> 13;
  int chain = g & 8191;
  float Av = -__expf(Alog[d * 16 + s]);
  float h = hin[(size_t)c * 8192 + chain];
  int l0 = c * CHLEN;
  for (int i = 0; i < CHLEN; ++i) {
    int l = l0 + i;
    if (i + 16 < CHLEN) {
      __builtin_prefetch(&delta[(size_t)(l + 16) * 512 + d], 0, 0);
      __builtin_prefetch(&u[(size_t)(l + 16) * 512 + d], 0, 0);
    }
    float dl = delta[(size_t)l * 512 + d];
    float dA = __expf(dl * Av);
    h = fmaf(dA, h, u[(size_t)l * 512 + d] * xdbl[(size_t)l * 48 + 16 + s]);
    float contrib = h * xdbl[(size_t)l * 48 + 32 + s];
    contrib += __shfl_xor(contrib, 8, 32);   // masks < 16: stays inside each
    contrib += __shfl_xor(contrib, 4, 32);   // 16-lane half (s = lane[3:0])
    contrib += __shfl_xor(contrib, 2, 32);
    contrib += __shfl_xor(contrib, 1, 32);
    if (s == 0) ycore[(size_t)l * 512 + d] = contrib;
  }
}

// y = (ycore + xi*D) * silu(z), cast to f16 for the output GEMM
__global__ __launch_bounds__(256) void k_yfin(const float* __restrict__ ycore,
                                              const float* __restrict__ xis,
                                              const float* __restrict__ Dp,
                                              const float* __restrict__ xz,
                                              _Float16* __restrict__ y16) {
  int g = blockIdx.x * blockDim.x + threadIdx.x;
  int d = g & (DIN - 1);
  int l = g >> 9;
  float y = fmaf(xis[g], Dp[d], ycore[g]);
  float z = xz[(size_t)l * 1024 + 512 + d];
  y *= z / (1.f + __expf(-z));
  y16[g] = (_Float16)y;
}

// ---------------- host-side orchestration ----------------
extern "C" void kernel_launch(void* const* d_in, const int* in_sizes, int n_in,
                              void* d_out, int out_size, void* d_ws, size_t ws_size,
                              hipStream_t stream) {
  const float* x     = (const float*)d_in[0];
  const float* gamma = (const float*)d_in[1];
  const float* beta  = (const float*)d_in[2];
  const float* Wo    = (const float*)d_in[21];
  const float* bo    = (const float*)d_in[22];
  float* out = (float*)d_out;

  char* ws = (char*)d_ws;
  _Float16* xn16   = (_Float16*)(ws + oXN16);
  _Float16* win16  = (_Float16*)(ws + oWIN16);
  _Float16* wx16   = (_Float16*)(ws + oWX16);
  _Float16* wout16 = (_Float16*)(ws + oWOUT16);
  _Float16* wo16   = (_Float16*)(ws + oWO16);
  _Float16* ycat   = (_Float16*)(ws + oYCAT);
  float*    xz     = (float*)   (ws + oXZ);
  float*    xis    = (float*)   (ws + oXIS);
  _Float16* xi16   = (_Float16*)(ws + oXI16);
  float*    xdbl   = (float*)   (ws + oXDBL);
  float*    delta  = (float*)   (ws + oDELTA);
  float*    u      = (float*)   (ws + oU);
  float*    ach    = (float*)   (ws + oACH);
  float*    bch    = (float*)   (ws + oBCH);
  float*    hin    = (float*)   (ws + oHIN);
  float*    ycore  = (float*)   (ws + oYCORE);
  _Float16* y16    = (_Float16*)(ws + oY16);

  k_ln<<<LSEQ / 256, 256, 0, stream>>>(x, gamma, beta, xn16);
  k_cvt<<<(256 * 512) / 256, 256, 0, stream>>>(Wo, wo16, 256 * 512);

  for (int b = 0; b < 2; ++b) {
    const float* Win   = (const float*)d_in[3 + b * 9 + 0];
    const float* convw = (const float*)d_in[3 + b * 9 + 1];
    const float* convb = (const float*)d_in[3 + b * 9 + 2];
    const float* Wx    = (const float*)d_in[3 + b * 9 + 3];
    const float* Wdt   = (const float*)d_in[3 + b * 9 + 4];
    const float* bdt   = (const float*)d_in[3 + b * 9 + 5];
    const float* Alog  = (const float*)d_in[3 + b * 9 + 6];
    const float* Dp    = (const float*)d_in[3 + b * 9 + 7];
    const float* Wout  = (const float*)d_in[3 + b * 9 + 8];

    k_cvt<<<(1024 * 256) / 256, 256, 0, stream>>>(Win, win16, 1024 * 256);
    k_cvt<<<(48 * 512) / 256, 256, 0, stream>>>(Wx, wx16, 48 * 512);
    k_cvt<<<(256 * 512) / 256, 256, 0, stream>>>(Wout, wout16, 256 * 512);

    // xz = xn(rev?) * Win^T : 128x16 blocks of 32x64 -> 2048 waves -> 256 blocks
    k_gemm1<<<256, 256, 0, stream>>>(xn16, win16, xz, b);
    // conv + silu
    k_conv<<<(LSEQ * DIN) / 256, 256, 0, stream>>>(xz, convw, convb, xis, xi16);
    // x_dbl = xi * Wx^T : 768 tiles -> 96 blocks
    k_gemm_xdbl<<<96, 256, 0, stream>>>(xi16, wx16, xdbl);
    // delta / u
    k_delta_u<<<(LSEQ * DIN) / 256, 256, 0, stream>>>(xdbl, Wdt, bdt, xis, delta, u);
    // chunked associative scan
    k_scan1<<<(8192 * NCHUNK) / 256, 256, 0, stream>>>(delta, u, xdbl, Alog, ach, bch);
    k_scan2<<<8192 / 256, 256, 0, stream>>>(ach, bch, hin);
    k_scan3<<<(8192 * NCHUNK) / 256, 256, 0, stream>>>(delta, u, xdbl, Alog, hin, ycore);
    // gate + D skip
    k_yfin<<<(LSEQ * DIN) / 256, 256, 0, stream>>>(ycore, xis, Dp, xz, y16);
    // branch output into concat buffer: 128x4 blocks -> 512 waves -> 64 blocks
    k_gemm_ycat<<<64, 256, 0, stream>>>(y16, wout16, ycat, b, b * 256);
  }

  // out = ycat * Wo^T + bo, stored (c, l)
  k_gemm_final<<<64, 256, 0, stream>>>(ycat, wo16, bo, out);
}
